// CrossTensorAttention_42597485642439
// MI455X (gfx1250) — matrix operand order repaired and checked by reading, not compiled, not run on hardware
//
#include <hip/hip_runtime.h>

typedef __attribute__((ext_vector_type(16))) _Float16 v16h;
typedef __attribute__((ext_vector_type(8)))  float    v8f;

// ---------------------------------------------------------------------------
// Generic batched WMMA GEMM:  C[M,N] (+)= A[M,K](f16) @ B[K,N](f16) + bias
// Row-major. Requires N%64==0, K%32==0. M guarded (row-clamped loads, masked
// stores). 256 threads = 8 waves; macro tile 128x64; each wave 32x32 via
// 2x2 fragments -> 4 v_wmma per K-step. A tile staged with
// GLOBAL_LOAD_ASYNC_TO_LDS_B128 (ASYNCcnt path), B staged with a transposing
// LDS store so both fragment reads are contiguous ds_load_b128s.
// ---------------------------------------------------------------------------
__global__ __launch_bounds__(256)
void gemm_wmma_kernel(const _Float16* __restrict__ A, const _Float16* __restrict__ B,
                      float* __restrict__ C, const float* __restrict__ bias,
                      int M, int N, int K,
                      long strideA, long strideB, long strideC, int accum)
{
    __shared__ _Float16 shA[128][48];   // [m][k], k padded to 48 (96B rows)
    __shared__ _Float16 shB[64][48];    // transposed: [n][k]

    const int bz = blockIdx.z;
    A += (long)bz * strideA;
    B += (long)bz * strideB;
    C += (long)bz * strideC;

    const int tid  = threadIdx.x;
    const int m0   = blockIdx.x * 128;
    const int n0   = blockIdx.y * 64;
    const int wv   = tid >> 5;
    const int lane = tid & 31;
    const int wm   = wv >> 1;          // 0..3  (32-row band)
    const int wn   = wv & 1;           // 0..1  (32-col band)
    const int half_id = lane >> 4;     // 0/1
    const int mrow    = lane & 15;

    v8f acc00 = {}, acc01 = {}, acc10 = {}, acc11 = {};

    const int arow = tid >> 2;              // 0..63 (two chunks: rows arow, arow+64)
    const int acol = (tid & 3) * 8;         // 0,8,16,24
    const int brow = tid >> 3;              // 0..31  (k)
    const int bcol = (tid & 7) * 8;         // 0..56  (n)

    // clamp A rows into range: rows >= M produce garbage that only lands in
    // C rows >= M, which are never stored.
    const int gr0 = (m0 + arow       < M) ? (m0 + arow)       : (M - 1);
    const int gr1 = (m0 + arow + 64  < M) ? (m0 + arow + 64)  : (M - 1);
    const unsigned ldsA0 = (unsigned)(unsigned long long)(const void*)&shA[arow][acol];
    const unsigned ldsA1 = (unsigned)(unsigned long long)(const void*)&shA[arow + 64][acol];

    for (int k0 = 0; k0 < K; k0 += 32) {
        // ---- A tile: async global -> LDS (128x32 halfs, 16B per lane x2) ----
        {
            unsigned long long ga0 = (unsigned long long)(const void*)(A + (long)gr0 * K + k0 + acol);
            unsigned long long ga1 = (unsigned long long)(const void*)(A + (long)gr1 * K + k0 + acol);
            asm volatile("global_load_async_to_lds_b128 %0, %1, off"
                         :: "v"(ldsA0), "v"(ga0) : "memory");
            asm volatile("global_load_async_to_lds_b128 %0, %1, off"
                         :: "v"(ldsA1), "v"(ga1) : "memory");
        }
        // ---- B tile: load + transpose into LDS ----
        const _Float16* bp = B + (long)(k0 + brow) * N + n0 + bcol;
        uint4 bv = *(const uint4*)bp;
        if (k0 + 32 < K) __builtin_prefetch(bp + 32L * N, 0, 1);
        const _Float16* bh = (const _Float16*)&bv;
        #pragma unroll
        for (int j = 0; j < 8; ++j) shB[bcol + j][brow] = bh[j];

        asm volatile("s_wait_asynccnt 0x0" ::: "memory");
        __syncthreads();

        // ---- fragments (per ISA 16-bit A/B VGPR layouts) ----
        v16h a0, a1, b0, b1;
        {
            const _Float16* p0  = &shA[wm * 32 + mrow][8 * half_id];
            const _Float16* p0b = &shA[wm * 32 + mrow][16 + 8 * half_id];
            const _Float16* p1  = &shA[wm * 32 + 16 + mrow][8 * half_id];
            const _Float16* p1b = &shA[wm * 32 + 16 + mrow][16 + 8 * half_id];
            #pragma unroll
            for (int j = 0; j < 8; ++j) {
                a0[j] = p0[j]; a0[8 + j] = p0b[j];
                a1[j] = p1[j]; a1[8 + j] = p1b[j];
            }
            const _Float16* q0  = &shB[wn * 32 + mrow][8 * half_id];
            const _Float16* q0b = &shB[wn * 32 + mrow][16 + 8 * half_id];
            const _Float16* q1  = &shB[wn * 32 + 16 + mrow][8 * half_id];
            const _Float16* q1b = &shB[wn * 32 + 16 + mrow][16 + 8 * half_id];
            #pragma unroll
            for (int j = 0; j < 8; ++j) {
                b0[j] = q0[j]; b0[8 + j] = q0b[j];
                b1[j] = q1[j]; b1[8 + j] = q1b[j];
            }
        }

        acc00 = __builtin_amdgcn_wmma_f32_16x16x32_f16(false, a0, false, b0, (short)0, acc00, false, false);
        acc01 = __builtin_amdgcn_wmma_f32_16x16x32_f16(false, a0, false, b1, (short)0, acc01, false, false);
        acc10 = __builtin_amdgcn_wmma_f32_16x16x32_f16(false, a1, false, b0, (short)0, acc10, false, false);
        acc11 = __builtin_amdgcn_wmma_f32_16x16x32_f16(false, a1, false, b1, (short)0, acc11, false, false);
        __syncthreads();
    }

    // ---- epilogue: C/D layout: col = lane%16, row = r + 8*(lane/16) ----
    #pragma unroll
    for (int ti = 0; ti < 2; ++ti) {
        #pragma unroll
        for (int tj = 0; tj < 2; ++tj) {
            v8f acc = ti ? (tj ? acc11 : acc10) : (tj ? acc01 : acc00);
            const int col = n0 + wn * 32 + tj * 16 + mrow;
            const float bvl = bias ? bias[col] : 0.0f;
            #pragma unroll
            for (int r = 0; r < 8; ++r) {
                const int row = m0 + wm * 32 + ti * 16 + half_id * 8 + r;
                if (row < M) {
                    const long idx = (long)row * N + col;
                    float v = acc[r] + bvl;
                    if (accum) v += C[idx];
                    C[idx] = v;
                }
            }
        }
    }
}

// ---------------------------------------------------------------------------
// Elementwise / normalization kernels
// ---------------------------------------------------------------------------
__global__ void f32_to_f16_kernel(const float* __restrict__ src, _Float16* __restrict__ dst, long n)
{
    long i = (long)blockIdx.x * blockDim.x + threadIdx.x;
    if (i < n) dst[i] = (_Float16)src[i];
}

__global__ void gelu_f16_kernel(const float* __restrict__ x, _Float16* __restrict__ out, long n)
{
    long i = (long)blockIdx.x * blockDim.x + threadIdx.x;
    if (i < n) {
        float v = x[i];
        out[i] = (_Float16)(0.5f * v * (1.0f + erff(v * 0.70710678118654752f)));
    }
}

__global__ __launch_bounds__(256)
void layernorm_f16_kernel(const float* __restrict__ x, const float* __restrict__ g,
                          const float* __restrict__ b, _Float16* __restrict__ out, int D)
{
    __shared__ float red[256];
    const int row = blockIdx.x;
    const float* xr = x + (long)row * D;
    float s = 0.f;
    for (int i = threadIdx.x; i < D; i += 256) s += xr[i];
    red[threadIdx.x] = s; __syncthreads();
    for (int o = 128; o > 0; o >>= 1) { if (threadIdx.x < o) red[threadIdx.x] += red[threadIdx.x + o]; __syncthreads(); }
    const float mean = red[0] / (float)D; __syncthreads();
    float v = 0.f;
    for (int i = threadIdx.x; i < D; i += 256) { float d = xr[i] - mean; v += d * d; }
    red[threadIdx.x] = v; __syncthreads();
    for (int o = 128; o > 0; o >>= 1) { if (threadIdx.x < o) red[threadIdx.x] += red[threadIdx.x + o]; __syncthreads(); }
    const float inv = rsqrtf(red[0] / (float)D + 1e-5f);
    _Float16* orow = out + (long)row * D;
    for (int i = threadIdx.x; i < D; i += 256)
        orow[i] = (_Float16)((xr[i] - mean) * inv * g[i] + b[i]);
}

// ---- reshape permutes (W==H==32) ----
__global__ void perm_c_kernel(const float* __restrict__ src, float* __restrict__ dst)
{   // dst[b][c][h*32+w] = src[b][c][w*32+h]   (b,c merged: 1024 rows)
    int i = blockIdx.x * 256 + threadIdx.x;
    if (i >= 4 * 256 * 1024) return;
    int s = i & 1023, bc = i >> 10;
    int w = s >> 5, h = s & 31;
    dst[(bc << 10) + (h << 5) + w] = src[i];
}
__global__ void perm_w_kernel(const float* __restrict__ src, float* __restrict__ dst)
{   // src (4,32,1024:s=w*32+h) -> dst[(b*32+w)*1024 + c*32 + h]
    int i = blockIdx.x * 256 + threadIdx.x;
    if (i >= 4 * 32 * 1024) return;
    int s = i & 1023, t = i >> 10;
    int c = t & 31, b = t >> 5;
    int w = s >> 5, h = s & 31;
    dst[(((b << 5) + w) << 10) + (c << 5) + h] = src[i];
}
__global__ void perm_h_kernel(const float* __restrict__ src, float* __restrict__ dst)
{   // src (4,32,1024:s=w*32+h) -> dst[(b*32+h)*1024 + c*32 + w]
    int i = blockIdx.x * 256 + threadIdx.x;
    if (i >= 4 * 32 * 1024) return;
    int s = i & 1023, t = i >> 10;
    int c = t & 31, b = t >> 5;
    int w = s >> 5, h = s & 31;
    dst[(((b << 5) + h) << 10) + (c << 5) + w] = src[i];
}

__global__ void pack_cat_kernel(const float* __restrict__ src, _Float16* __restrict__ cat,
                                int rows, int rowoff)
{   // src (4, rows, 1024) -> cat (4,512,1024) rows [rowoff, rowoff+rows)
    long i = (long)blockIdx.x * 256 + threadIdx.x;
    long total = 4L * rows * 1024;
    if (i >= total) return;
    int col = (int)(i & 1023);
    long t = i >> 10;
    int r = (int)(t % rows), b = (int)(t / rows);
    cat[((long)b * 512 + rowoff + r) * 1024 + col] = (_Float16)src[i];
}

__global__ void init_scalar_kernel(unsigned int* s)
{
    if (blockIdx.x == 0 && threadIdx.x == 0) *s = 0u;
}

// ---------------------------------------------------------------------------
// Nystrom attention core. qkv rows: r = b*n + i, cols: q[0:128], k[128:256], v[256:384]
// ---------------------------------------------------------------------------
__global__ __launch_bounds__(256)
void attn_prep_kernel(const float* __restrict__ qkv, float* __restrict__ a2buf,
                      unsigned int* __restrict__ scal, int n, int l)
{
    const int blk = blockIdx.x;        // b*8 + h
    const int h = blk & 7, b = blk >> 3;
    const long base = (long)(b * n) * 384;
    __shared__ float ql[32][16], kl[32][16], a2[32][32], red[32];
    const int tid = threadIdx.x;

    for (int e = tid; e < 512; e += 256) {
        int j = e >> 4, d = e & 15;
        float sq = 0.f, sk = 0.f;
        for (int u = 0; u < l; ++u) {
            long r = base + (long)(j * l + u) * 384;
            sq += qkv[r + h * 16 + d];
            sk += qkv[r + 128 + h * 16 + d];
        }
        ql[j][d] = sq * 0.25f / (float)l;
        kl[j][d] = sk / (float)l;
    }
    __syncthreads();
    for (int e = tid; e < 1024; e += 256) {
        int i = e >> 5, j = e & 31;
        float s = 0.f;
        for (int d = 0; d < 16; ++d) s += ql[i][d] * kl[j][d];
        a2[i][j] = s;
    }
    __syncthreads();
    if (tid < 32) {
        float mx = -1e30f;
        for (int j = 0; j < 32; ++j) mx = fmaxf(mx, a2[tid][j]);
        float sum = 0.f;
        for (int j = 0; j < 32; ++j) { float e = expf(a2[tid][j] - mx); a2[tid][j] = e; sum += e; }
        float inv = 1.0f / sum;
        for (int j = 0; j < 32; ++j) a2[tid][j] *= inv;
    }
    __syncthreads();
    for (int e = tid; e < 1024; e += 256) a2buf[(long)blk * 1024 + e] = a2[e >> 5][e & 31];
    if (tid < 32) {
        float cs = 0.f;
        for (int i = 0; i < 32; ++i) cs += a2[i][tid];
        red[tid] = cs;
    }
    __syncthreads();
    if (tid == 0) {
        float mx = 0.f;
        for (int j = 0; j < 32; ++j) mx = fmaxf(mx, red[j]);
        atomicMax(scal, __float_as_uint(mx));   // positive floats: uint order == float order
    }
}

__global__ __launch_bounds__(256)
void attn_main_kernel(const float* __restrict__ qkv, const float* __restrict__ a2buf,
                      const unsigned int* __restrict__ scal, const float* __restrict__ resk,
                      _Float16* __restrict__ out, int n, int l)
{
    const int blk = blockIdx.x;
    const int h = blk & 7, b = blk >> 3;
    const long base = (long)(b * n) * 384;
    __shared__ float sk[256 * 16];
    __shared__ float sv[256 * 16];
    __shared__ float sa3[32 * 256];
    __shared__ float a2[1024], z[1024], az[1024], m1[1024], m2[1024];
    __shared__ float ql[32][16], kl[32][16], kv[32][16];
    __shared__ float rk[33];
    const int tid = threadIdx.x;

    for (int e = tid; e < n * 16; e += 256) {
        int i = e >> 4, d = e & 15;
        long r = base + (long)i * 384 + h * 16;
        sk[e] = qkv[r + 128 + d];
        sv[e] = qkv[r + 256 + d];
    }
    if (tid < 33) rk[tid] = resk[h * 33 + tid];
    __syncthreads();

    for (int e = tid; e < 512; e += 256) {
        int j = e >> 4, d = e & 15;
        float sq = 0.f, skk = 0.f;
        for (int u = 0; u < l; ++u) {
            int i = j * l + u;
            sq  += qkv[base + (long)i * 384 + h * 16 + d];
            skk += sk[i * 16 + d];
        }
        ql[j][d] = sq * 0.25f / (float)l;
        kl[j][d] = skk / (float)l;
    }
    __syncthreads();

    const float dinv = 1.0f / __uint_as_float(*scal);   // row-sum max of softmax == 1
    for (int e = tid; e < 1024; e += 256) {
        float v = a2buf[(long)blk * 1024 + e];
        a2[e] = v;
        int i = e >> 5, j = e & 31;
        z[j * 32 + i] = v * dinv;                        // z = a2^T / denom
    }
    __syncthreads();

    // Newton-Schulz pinv, 6 iterations: z = 0.25 z (13I - az(15I - az(7I - az)))
    for (int it = 0; it < 6; ++it) {
        for (int e = tid; e < 1024; e += 256) { int i = e >> 5, j = e & 31; float s = 0.f;
            for (int p = 0; p < 32; ++p) s += a2[i * 32 + p] * z[p * 32 + j]; az[e] = s; }
        __syncthreads();
        for (int e = tid; e < 1024; e += 256) { int i = e >> 5, j = e & 31; m1[e] = (i == j ? 7.0f : 0.0f) - az[e]; }
        __syncthreads();
        for (int e = tid; e < 1024; e += 256) { int i = e >> 5, j = e & 31; float s = 0.f;
            for (int p = 0; p < 32; ++p) s += az[i * 32 + p] * m1[p * 32 + j]; m2[e] = s; }
        __syncthreads();
        for (int e = tid; e < 1024; e += 256) { int i = e >> 5, j = e & 31; m1[e] = (i == j ? 15.0f : 0.0f) - m2[e]; }
        __syncthreads();
        for (int e = tid; e < 1024; e += 256) { int i = e >> 5, j = e & 31; float s = 0.f;
            for (int p = 0; p < 32; ++p) s += az[i * 32 + p] * m1[p * 32 + j]; m2[e] = s; }
        __syncthreads();
        for (int e = tid; e < 1024; e += 256) { int i = e >> 5, j = e & 31; m1[e] = (i == j ? 13.0f : 0.0f) - m2[e]; }
        __syncthreads();
        for (int e = tid; e < 1024; e += 256) { int i = e >> 5, j = e & 31; float s = 0.f;
            for (int p = 0; p < 32; ++p) s += z[i * 32 + p] * m1[p * 32 + j]; m2[e] = 0.25f * s; }
        __syncthreads();
        for (int e = tid; e < 1024; e += 256) z[e] = m2[e];
        __syncthreads();
    }

    // a3 = softmax(q_l @ k^T) rows of length n
    if (tid < 32) {
        const int j = tid;
        float mx = -1e30f;
        for (int i = 0; i < n; ++i) {
            float s = 0.f;
            for (int d = 0; d < 16; ++d) s += ql[j][d] * sk[i * 16 + d];
            sa3[j * n + i] = s;
            if (s > mx) mx = s;
        }
        float sum = 0.f;
        for (int i = 0; i < n; ++i) { float e = expf(sa3[j * n + i] - mx); sa3[j * n + i] = e; sum += e; }
        float inv = 1.0f / sum;
        for (int i = 0; i < n; ++i) sa3[j * n + i] *= inv;
    }
    __syncthreads();

    // kv = a3 @ v  (32 x 16)
    for (int e = tid; e < 512; e += 256) {
        int j = e >> 4, d = e & 15;
        float s = 0.f;
        for (int i = 0; i < n; ++i) s += sa3[j * n + i] * sv[i * 16 + d];
        kv[j][d] = s;
    }
    __syncthreads();

    // per-row: out = (softmax(q @ k_l^T) @ z) @ kv + depthwise-conv33(v)
    if (tid < n) {
        const int i = tid;
        float qrow[16];
        for (int d = 0; d < 16; ++d) qrow[d] = qkv[base + (long)i * 384 + h * 16 + d] * 0.25f;
        float a1r[32]; float mx = -1e30f;
        for (int j = 0; j < 32; ++j) {
            float s = 0.f;
            for (int d = 0; d < 16; ++d) s += qrow[d] * kl[j][d];
            a1r[j] = s; if (s > mx) mx = s;
        }
        float sum = 0.f;
        for (int j = 0; j < 32; ++j) { float e = expf(a1r[j] - mx); a1r[j] = e; sum += e; }
        float inv = 1.0f / sum;
        for (int j = 0; j < 32; ++j) a1r[j] *= inv;
        float o1[32];
        for (int j = 0; j < 32; ++j) {
            float s = 0.f;
            for (int p = 0; p < 32; ++p) s += a1r[p] * z[p * 32 + j];
            o1[j] = s;
        }
        for (int d = 0; d < 16; ++d) {
            float s = 0.f;
            for (int j = 0; j < 32; ++j) s += o1[j] * kv[j][d];
            float r = 0.f;
            for (int t = 0; t < 33; ++t) {
                int ii = i + t - 16;
                if (ii >= 0 && ii < n) r += sv[ii * 16 + d] * rk[t];
            }
            out[((long)(b * n + i)) * 128 + h * 16 + d] = (_Float16)(s + r);
        }
    }
}

// ---------------------------------------------------------------------------
// BatchNorm (training stats) + ReLU + AvgPool2x2 + ReLU
// ---------------------------------------------------------------------------
__global__ __launch_bounds__(256)
void bn_stats_kernel(const float* __restrict__ y, float* __restrict__ stat)
{   // y (4,128,1024); stat[0:128]=mean, stat[128:256]=invstd
    const int o = blockIdx.x;
    __shared__ float r1[256], r2[256];
    float s = 0.f, s2 = 0.f;
    for (int e = threadIdx.x; e < 4096; e += 256) {
        int b = e >> 10, sp = e & 1023;
        float v = y[((long)b * 128 + o) * 1024 + sp];
        s += v; s2 += v * v;
    }
    r1[threadIdx.x] = s; r2[threadIdx.x] = s2; __syncthreads();
    for (int off = 128; off > 0; off >>= 1) {
        if (threadIdx.x < off) { r1[threadIdx.x] += r1[threadIdx.x + off]; r2[threadIdx.x] += r2[threadIdx.x + off]; }
        __syncthreads();
    }
    if (threadIdx.x == 0) {
        float m = r1[0] * (1.0f / 4096.0f);
        float var = r2[0] * (1.0f / 4096.0f) - m * m;
        stat[o] = m;
        stat[128 + o] = rsqrtf(var + 1e-5f);
    }
}

__global__ void bn_pool_kernel(const float* __restrict__ y, const float* __restrict__ stat,
                               const float* __restrict__ g, const float* __restrict__ bb,
                               float* __restrict__ out)
{
    int idx = blockIdx.x * 256 + threadIdx.x;      // (4,128,16,16) = 131072
    if (idx >= 131072) return;
    int j = idx & 15, i = (idx >> 4) & 15, o = (idx >> 8) & 127, b = idx >> 15;
    float m = stat[o], inv = stat[128 + o], gg = g[o], bs = bb[o];
    float acc = 0.f;
    for (int di = 0; di < 2; ++di)
        for (int dj = 0; dj < 2; ++dj) {
            int s = (2 * i + di) * 32 + 2 * j + dj;
            float v = y[((long)b * 128 + o) * 1024 + s];
            v = (v - m) * inv * gg + bs;
            acc += fmaxf(v, 0.f);
        }
    acc *= 0.25f;
    out[idx] = fmaxf(acc, 0.f);
}

// ---------------------------------------------------------------------------
// Host orchestration
// ---------------------------------------------------------------------------
static inline void launch_gemm(hipStream_t st, const _Float16* A, const _Float16* B, float* C,
                               const float* bias, int M, int N, int K,
                               long sa, long sb, long sc, int batch, int accum)
{
    dim3 grid((M + 127) / 128, N / 64, batch);
    gemm_wmma_kernel<<<grid, 256, 0, st>>>(A, B, C, bias, M, N, K, sa, sb, sc, accum);
}
static inline void launch_cvt(hipStream_t st, const float* s, _Float16* d, long n)
{
    f32_to_f16_kernel<<<dim3((unsigned)((n + 255) / 256)), 256, 0, st>>>(s, d, n);
}

static void run_nystromformer(hipStream_t st, const float* const* in, int base_idx,
                              float* seq, int n,
                              float* t0, float* a2buf, unsigned int* scal,
                              _Float16* hA, _Float16* hB)
{
    const int rows = 4 * n;
    const int l = n / 32;
    for (int L = 0; L < 4; ++L) {
        const float* ln1_g = in[base_idx + L * 12 + 0];
        const float* ln1_b = in[base_idx + L * 12 + 1];
        const float* w_qkv = in[base_idx + L * 12 + 2];
        const float* res_k = in[base_idx + L * 12 + 3];
        const float* w_out = in[base_idx + L * 12 + 4];
        const float* b_out = in[base_idx + L * 12 + 5];
        const float* ln2_g = in[base_idx + L * 12 + 6];
        const float* ln2_b = in[base_idx + L * 12 + 7];
        const float* w_ff1 = in[base_idx + L * 12 + 8];
        const float* b_ff1 = in[base_idx + L * 12 + 9];
        const float* w_ff2 = in[base_idx + L * 12 + 10];
        const float* b_ff2 = in[base_idx + L * 12 + 11];

        // attention block
        layernorm_f16_kernel<<<rows, 256, 0, st>>>(seq, ln1_g, ln1_b, hA, 1024);
        launch_cvt(st, w_qkv, hB, 1024L * 384);
        launch_gemm(st, hA, hB, t0, nullptr, rows, 384, 1024, 0, 0, 0, 1, 0);   // qkv
        init_scalar_kernel<<<1, 32, 0, st>>>(scal);
        attn_prep_kernel<<<32, 256, 0, st>>>(t0, a2buf, scal, n, l);
        attn_main_kernel<<<32, 256, 0, st>>>(t0, a2buf, scal, res_k, hA, n, l);
        launch_cvt(st, w_out, hB, 128L * 1024);
        launch_gemm(st, hA, hB, seq, b_out, rows, 1024, 128, 0, 0, 0, 1, 1);    // out proj += residual

        // feed-forward block
        layernorm_f16_kernel<<<rows, 256, 0, st>>>(seq, ln2_g, ln2_b, hA, 1024);
        launch_cvt(st, w_ff1, hB, 1024L * 4096);
        launch_gemm(st, hA, hB, t0, b_ff1, rows, 4096, 1024, 0, 0, 0, 1, 0);
        {
            long nn = (long)rows * 4096;
            gelu_f16_kernel<<<dim3((unsigned)((nn + 255) / 256)), 256, 0, st>>>(t0, hA, nn);
        }
        launch_cvt(st, w_ff2, hB, 4096L * 1024);
        launch_gemm(st, hA, hB, seq, b_ff2, rows, 1024, 4096, 0, 0, 0, 1, 1);
    }
}

extern "C" void kernel_launch(void* const* d_in, const int* in_sizes, int n_in,
                              void* d_out, int out_size, void* d_ws, size_t ws_size,
                              hipStream_t stream)
{
    const float* const* in = (const float* const*)d_in;

    // input index map (setup_inputs dict order, nested dicts/lists flattened)
    const float* x    = in[0];
    const float* c_w  = in[1]; const float* c_b  = in[2];
    const float* w_w  = in[3]; const float* w_b  = in[4];
    const float* h_w  = in[5]; const float* h_b  = in[6];
    const int nys_c0 = 7, nys_w0 = 7 + 48, nys_h0 = 7 + 96;
    const float* incw_w = in[151]; const float* incw_b = in[152];
    const float* inch_w = in[153]; const float* inch_b = in[154];
    const float* fin_w  = in[155]; const float* fin_b  = in[156];
    const float* bn_g   = in[157]; const float* bn_b   = in[158];

    // workspace layout (float offsets, all 256B aligned)
    float* wsf = (float*)d_ws;
    float* seq_c = wsf;                       // 1,048,576
    float* seq_w = wsf + 1048576;             //   131,072
    float* seq_h = wsf + 1179648;             //   131,072
    float* t0    = wsf + 1310720;             // 4,194,304
    float* wred  = wsf + 5505024;             //   524,288
    float* hred  = wsf + 6029312;             //   524,288
    float* ybuf  = wsf + 6553600;             //   524,288
    float* a2buf = wsf + 7077888;             //    32,768
    float* stat  = wsf + 7110656;             //       256
    unsigned int* scal = (unsigned int*)(wsf + 7110912);
    _Float16* hA = (_Float16*)((char*)d_ws + 28443904);   // 4,194,304 halfs
    _Float16* hB = hA + 4194304;                          // 4,194,304 halfs

    hipStream_t st = stream;

    // ---- stage 1: 1x1 convs + reshapes ----
    launch_cvt(st, x, hB, 1048576);                       // x as f16 (B operand, per-batch 256x1024)
    launch_cvt(st, c_w, hA, 65536);
    launch_gemm(st, hA, hB, t0, c_b, 256, 1024, 256, 0, 262144, 262144, 4, 0);
    perm_c_kernel<<<4096, 256, 0, st>>>(t0, seq_c);
    launch_cvt(st, w_w, hA, 8192);
    launch_gemm(st, hA, hB, t0, w_b, 32, 1024, 256, 0, 262144, 32768, 4, 0);
    perm_w_kernel<<<512, 256, 0, st>>>(t0, seq_w);
    launch_cvt(st, h_w, hA, 8192);
    launch_gemm(st, hA, hB, t0, h_b, 32, 1024, 256, 0, 262144, 32768, 4, 0);
    perm_h_kernel<<<512, 256, 0, st>>>(t0, seq_h);

    // ---- stage 2: three nystromformers ----
    run_nystromformer(st, in, nys_c0, seq_c, 256, t0, a2buf, scal, hA, hB);
    run_nystromformer(st, in, nys_w0, seq_w, 32, t0, a2buf, scal, hA, hB);
    run_nystromformer(st, in, nys_h0, seq_h, 32, t0, a2buf, scal, hA, hB);

    // ---- stage 3: channel-increase projections (batched, weight as A) ----
    launch_cvt(st, incw_w, hA, 4096);
    launch_cvt(st, seq_w, hB, 131072);
    launch_gemm(st, hA, hB, wred, incw_b, 128, 1024, 32, 0, 32768, 131072, 4, 0);
    launch_cvt(st, inch_w, hA, 4096);
    launch_cvt(st, seq_h, hB, 131072);
    launch_gemm(st, hA, hB, hred, inch_b, 128, 1024, 32, 0, 32768, 131072, 4, 0);

    // ---- stage 4: concat -> final 1x1 conv -> BN -> ReLU -> pool -> ReLU ----
    pack_cat_kernel<<<4096, 256, 0, st>>>(seq_c, hB, 256, 0);
    pack_cat_kernel<<<2048, 256, 0, st>>>(wred,  hB, 128, 256);
    pack_cat_kernel<<<2048, 256, 0, st>>>(hred,  hB, 128, 384);
    launch_cvt(st, fin_w, hA, 65536);
    launch_gemm(st, hA, hB, ybuf, fin_b, 128, 1024, 512, 0, 524288, 131072, 4, 0);
    bn_stats_kernel<<<128, 256, 0, st>>>(ybuf, stat);
    bn_pool_kernel<<<512, 256, 0, st>>>(ybuf, stat, bn_g, bn_b, (float*)d_out);

    (void)in_sizes; (void)n_in; (void)out_size; (void)ws_size;
}
